// DTD_17557826306396
// MI455X (gfx1250) — matrix operand order repaired and checked by reading, not compiled, not run on hardware
//
#include <hip/hip_runtime.h>

// ---------------------------------------------------------------------------
// Problem constants (fixed by setup_inputs)
// ---------------------------------------------------------------------------
#define NB   128            // batch
#define NC   512            // channels
#define HW   512            // nH*nW = 16*32
#define NT   40             // time steps (>= nsteps, safe upper bound)
#define HID  256            // LSTM hidden per direction
#define G4   1024           // 4*HID  (LSTM gates)
#define NX   1024           // 2*NC   (GRU input = concat(C2, emb))
#define G3   1536           // 3*NC   (GRU gates)
#define NCL  5000           // classes
#define NTL  320            // generator N tiles padded to multiple of 8 (313 live)
#define M5   (NT * NB)      // 5120 flattened (t,b) rows

typedef __attribute__((ext_vector_type(16))) _Float16 v16h;
typedef __attribute__((ext_vector_type(8)))  _Float16 v8h;
typedef __attribute__((ext_vector_type(4)))  _Float16 v4h;
typedef __attribute__((ext_vector_type(8)))  float    v8f;
typedef __attribute__((ext_vector_type(4)))  unsigned int u32x4;
typedef __attribute__((ext_vector_type(4)))  int i32x4;
typedef __attribute__((ext_vector_type(8)))  int i32x8;

#if defined(__has_builtin)
#if __has_builtin(__builtin_amdgcn_tensor_load_to_lds)
#define HAVE_TDM 1
#endif
#endif

__device__ __forceinline__ float sigm(float x) { return 1.0f / (1.0f + __expf(-x)); }

// D = A(16x32 f16) * B(32x16 f16) + C(f32)  -- CDNA5 WMMA, wave32
__device__ __forceinline__ v8f wmma16(v16h a, v16h b, v8f c) {
  return __builtin_amdgcn_wmma_f32_16x16x32_f16(false, a, false, b, (short)0, c,
                                                false, false);
}

// ---------------------------------------------------------------------------
// f16 fragment loaders (operands pre-converted; K pairs contiguous -> b128s).
// ---------------------------------------------------------------------------
__device__ __forceinline__ v16h cat16(v8h lo, v8h hi) {
  return __builtin_shufflevector(lo, hi, 0, 1, 2, 3, 4, 5, 6, 7,
                                 8, 9, 10, 11, 12, 13, 14, 15);
}

__device__ __forceinline__ v16h load_a16h(const _Float16* __restrict__ act, int lda,
                                          int row0, int rowMax, int k0) {
  const int lane = threadIdx.x & 31;
  const int half = lane >> 4, m = lane & 15;
  int row = row0 + m; if (row > rowMax) row = rowMax;
  const _Float16* rp = act + (size_t)row * lda + k0 + 16 * half;
  return cat16(*(const v8h*)rp, *(const v8h*)(rp + 8));
}

__device__ __forceinline__ v16h load_bw16h(const _Float16* __restrict__ W, int ldw,
                                           int col0, int colMax, int k0) {
  const int lane = threadIdx.x & 31;
  const int half = lane >> 4, n = lane & 15;
  int col = col0 + n; if (col > colMax) col = colMax;
  const _Float16* rp = W + (size_t)col * ldw + k0 + 16 * half;
  return cat16(*(const v8h*)rp, *(const v8h*)(rp + 8));
}

// f32 B loader (only for streaming `feature`, not worth pre-converting)
__device__ __forceinline__ v16h load_bw32(const float* __restrict__ W, int ldw,
                                          int col0, int colMax, int k0) {
  const int lane = threadIdx.x & 31;
  const int half = lane >> 4, n = lane & 15;
  int col = col0 + n; if (col > colMax) col = colMax;
  const float* rp = W + (size_t)col * ldw + k0 + 16 * half;
  v16h b;
#pragma unroll
  for (int v = 0; v < 8; ++v) {
    b[2 * v]     = (_Float16)rp[2 * v];
    b[2 * v + 1] = (_Float16)rp[2 * v + 1];
  }
  return b;
}

// A fragment out of an LDS-staged 16xKD tile -> ds_load_b128 pairs
__device__ __forceinline__ v16h lds_a16(const _Float16* tile, int kd, int k0) {
  const int lane = threadIdx.x & 31;
  const int half = lane >> 4, m = lane & 15;
  const _Float16* rp = tile + m * kd + k0 + 16 * half;
  return cat16(*(const v8h*)rp, *(const v8h*)(rp + 8));
}

// Manual cooperative stage of a 16xKD f16 A-tile into LDS (256 threads)
__device__ __forceinline__ void stage_a16(_Float16* tile, const _Float16* src,
                                          int kd, int row0) {
  const int cpr = kd >> 3;              // b128 chunks per row
  for (int c = threadIdx.x; c < 16 * cpr; c += 256) {
    int row = c / cpr;
    int off = (c - row * cpr) << 3;
    *(v8h*)(tile + row * kd + off) =
        *(const v8h*)(src + (size_t)(row0 + row) * kd + off);
  }
  __syncthreads();
}

// ---------------------------------------------------------------------------
// TDM stage: one wave issues a Tensor Data Mover descriptor that DMAs the
// 16 x kd f16 tile (row stride kd) from global into LDS, waits on TENSORcnt,
// then the workgroup barrier releases the consumer waves.
// D# layout per CDNA5 ISA ch.8: group0 = count/lds_addr/global_addr/type,
// group1 = data_size + tensor/tile dims + dim0 stride, groups 2-3 zero (2D).
// ---------------------------------------------------------------------------
__device__ __forceinline__ void stage_tdm(_Float16* tile, const _Float16* src,
                                          int kd, int row0) {
#ifdef HAVE_TDM
  if (threadIdx.x < 32) {               // wave 0 only; TDM ignores EXEC
    unsigned long long ga =
        (unsigned long long)(const void*)(src + (size_t)row0 * kd);
    unsigned lds = (unsigned)(unsigned long long)(void*)tile;  // addr[31:0] = LDS byte offset
    u32x4 g0;
    g0[0] = 1u;                                    // count=1 valid descriptor
    g0[1] = lds;                                   // lds_addr (bytes)
    g0[2] = (unsigned)(ga & 0xFFFFFFFFu);          // global_addr[31:0]
    g0[3] = (unsigned)((ga >> 32) & 0x01FFFFFFu)   // global_addr[56:32]
          | (2u << 30);                            // type = 2
    i32x8 g1;
    g1[0] = (int)(1u << 16);                       // wg_mask=0, data_size=1 (2B)
    g1[1] = (int)(((unsigned)kd & 0xFFFFu) << 16); // tensor_dim0[15:0]
    g1[2] = (int)((((unsigned)kd >> 16) & 0xFFFFu) // tensor_dim0[31:16]
          | (16u << 16));                          // tensor_dim1 = 16 rows
    g1[3] = (int)(((unsigned)kd & 0xFFFFu) << 16); // tile_dim0 = kd
    g1[4] = 16;                                    // tile_dim1 = 16 rows
    g1[5] = kd;                                    // tensor_dim0_stride[31:0]
    g1[6] = 0;                                     // stride hi / dim1_stride lo
    g1[7] = 0;
    i32x4 gz = {0, 0, 0, 0};
#if __clang_major__ >= 23
    i32x8 gz8 = {0, 0, 0, 0, 0, 0, 0, 0};
    __builtin_amdgcn_tensor_load_to_lds(g0, g1, gz, gz, gz8, 0);
#else
    __builtin_amdgcn_tensor_load_to_lds(g0, g1, gz, gz, 0);
#endif
    __builtin_amdgcn_s_wait_tensorcnt(0);
  }
  __syncthreads();
#else
  stage_a16(tile, src, kd, row0);
#endif
}

// Same as stage_a16 but f32 source converted to f16 while staging
__device__ __forceinline__ void stage_a32(_Float16* tile, const float* src,
                                          int kd, int row0, int rowMax) {
  const int cpr = kd >> 2;              // float4 chunks per row
  for (int c = threadIdx.x; c < 16 * cpr; c += 256) {
    int row = c / cpr;
    int off = (c - row * cpr) << 2;
    int r = row0 + row; if (r > rowMax) r = rowMax;
    const float* s = src + (size_t)r * kd + off;
    v4h h;
    h[0] = (_Float16)s[0]; h[1] = (_Float16)s[1];
    h[2] = (_Float16)s[2]; h[3] = (_Float16)s[3];
    *(v4h*)(tile + row * kd + off) = h;
  }
  __syncthreads();
}

// ---------------------------------------------------------------------------
// one-time f32 -> f16 weight conversion
// ---------------------------------------------------------------------------
__global__ void k_cvt(const float* __restrict__ src, _Float16* __restrict__ dst,
                      int n4) {
  int i = blockIdx.x * blockDim.x + threadIdx.x;
  int stride = gridDim.x * blockDim.x;
  for (; i < n4; i += stride) {
    const float* s = src + 4 * (size_t)i;
    v4h h;
    h[0] = (_Float16)s[0]; h[1] = (_Float16)s[1];
    h[2] = (_Float16)s[2]; h[3] = (_Float16)s[3];
    *(v4h*)(dst + 4 * (size_t)i) = h;
  }
}

// K0: per-(b,t) attention normalizer (reciprocal)
__global__ void k_att_sums(const float* __restrict__ A, float* __restrict__ Sinv) {
  int w = blockIdx.x * (blockDim.x >> 5) + (threadIdx.x >> 5);
  if (w >= NB * NT) return;
  int lane = threadIdx.x & 31;
  const float* p = A + (size_t)w * HW;
  float s = 0.f;
  for (int i = lane; i < HW; i += 32) s += p[i];
#pragma unroll
  for (int m = 16; m; m >>= 1) s += __shfl_xor(s, m, 32);
  if (lane == 0) Sinv[w] = 1.0f / s;
}

// K0b: exclusive prefix sum of text lengths
__global__ void k_offsets(const int* __restrict__ tl, int* __restrict__ offs) {
  if (threadIdx.x == 0 && blockIdx.x == 0) {
    int acc = 0;
    for (int b = 0; b < NB; ++b) { offs[b] = acc; acc += tl[b]; }
  }
}

// ---------------------------------------------------------------------------
// K1: attention pooling. Per b: [40 x HW] x [HW x NC] -> Cpool16[t][b][c].
// Block's 8 waves share (b, mt): A tile staged to LDS (converted to f16).
// ---------------------------------------------------------------------------
__global__ void __launch_bounds__(256)
k_pool(const float* __restrict__ A, const float* __restrict__ feat,
       const float* __restrict__ Sinv, _Float16* __restrict__ Cpool16) {
  __shared__ _Float16 atile[16 * HW];   // 16 KB
  const int wib = threadIdx.x >> 5;
  const int w0 = blockIdx.x * 8;
  const int b = w0 / 96, rem0 = w0 % 96;
  const int mt = rem0 >> 5, nt = (rem0 & 31) + wib;
  const int lane = threadIdx.x & 31, half = lane >> 4, n16 = lane & 15;

  stage_a32(atile, A + (size_t)b * NT * HW, HW, mt * 16, NT - 1);

  const float* Fb = feat + (size_t)b * NC * HW;
  v8f acc; for (int r = 0; r < 8; ++r) acc[r] = 0.f;
  for (int k0 = 0; k0 < HW; k0 += 32) {
    v16h a  = lds_a16(atile, HW, k0);
    v16h bb = load_bw32(Fb, HW, nt * 16, NC - 1, k0);
    acc = wmma16(a, bb, acc);
  }
#pragma unroll
  for (int r = 0; r < 8; ++r) {
    int t = mt * 16 + r + 8 * half;
    if (t < NT) {
      int c = nt * 16 + n16;
      Cpool16[((size_t)t * NB + b) * NC + c] =
          (_Float16)(acc[r] * Sinv[b * NT + t]);
    }
  }
}

// ---------------------------------------------------------------------------
// K2: LSTM input projections, both dirs. [M5 x NC] x [NC x G4] + bih
// ---------------------------------------------------------------------------
__global__ void __launch_bounds__(256)
k_lstm_xg(const _Float16* __restrict__ Cpool16,
          const _Float16* __restrict__ Wf16, const float* __restrict__ biasf,
          const _Float16* __restrict__ Wb16, const float* __restrict__ biasb,
          float* __restrict__ xgf, float* __restrict__ xgb) {
  __shared__ _Float16 atile[16 * NC];   // 16 KB
  const int wib = threadIdx.x >> 5;
  const int w0 = blockIdx.x * 8;        // 0..40952
  const int dir = (w0 >= 320 * 64);
  const int r20 = dir ? w0 - 320 * 64 : w0;
  const int mt = r20 >> 6, nt = (r20 & 63) + wib;
  const _Float16* W  = dir ? Wb16 : Wf16;
  const float* bias  = dir ? biasb : biasf;
  float* xg          = dir ? xgb : xgf;
  const int lane = threadIdx.x & 31, half = lane >> 4, n16 = lane & 15;

  stage_tdm(atile, Cpool16, NC, mt * 16);

  v8f acc; for (int r = 0; r < 8; ++r) acc[r] = 0.f;
  for (int k0 = 0; k0 < NC; k0 += 32) {
    v16h a = lds_a16(atile, NC, k0);
    v16h b = load_bw16h(W, NC, nt * 16, G4 - 1, k0);
    acc = wmma16(a, b, acc);
  }
#pragma unroll
  for (int r = 0; r < 8; ++r) {
    int m = mt * 16 + r + 8 * half;
    int col = nt * 16 + n16;
    xg[(size_t)m * G4 + col] = acc[r] + bias[col];
  }
}

// ---------------------------------------------------------------------------
// K3: teacher-forcing embeddings -> xcat16[t][b][512:1024]
// ---------------------------------------------------------------------------
__global__ void k_emb(const int* __restrict__ text, const int* __restrict__ tl,
                      const int* __restrict__ offs, const float* __restrict__ emb,
                      _Float16* __restrict__ xcat16, int total) {
  int tb = blockIdx.x;                 // 0..M5-1
  int t = tb >> 7, b = tb & 127;
  int tok = 0;
  if (t >= 2) {
    int p = t - 2;
    if (p < tl[b]) {
      int idx = offs[b] + p; if (idx > total - 1) idx = total - 1;
      tok = text[idx] + 1;
    }
  }
  const float* erow = emb + (size_t)tok * NC;
  _Float16* dst = xcat16 + ((size_t)t * NB + b) * NX + NC;
  for (int c = threadIdx.x; c < NC; c += blockDim.x) dst[c] = (_Float16)erow[c];
}

// ---------------------------------------------------------------------------
// K4: fused LSTM step, both directions. 4 WMMA accumulators = gates i,f,g,o.
// ---------------------------------------------------------------------------
__global__ void __launch_bounds__(256)
k_lstm_step(const float* __restrict__ xgf, const float* __restrict__ xgb,
            const _Float16* __restrict__ Whf16, const _Float16* __restrict__ Whb16,
            const float* __restrict__ bhf, const float* __restrict__ bhb,
            const _Float16* __restrict__ hf_in, _Float16* __restrict__ hf_out,
            float* __restrict__ cf,
            const _Float16* __restrict__ hb_in, _Float16* __restrict__ hb_out,
            float* __restrict__ cb,
            _Float16* __restrict__ xcat16, int s) {
  const int dir = blockIdx.y;
  const int tt = dir ? (NT - 1 - s) : s;
  const float* xg        = dir ? xgb : xgf;
  const _Float16* Whh    = dir ? Whb16 : Whf16;
  const float* bhh       = dir ? bhb : bhf;
  const _Float16* hin    = dir ? hb_in : hf_in;
  _Float16* hout         = dir ? hb_out : hf_out;
  float* cst             = dir ? cb : cf;

  int w = blockIdx.x * 8 + (threadIdx.x >> 5);   // 0..127
  int bt = w >> 4, jt = w & 15;
  int lane = threadIdx.x & 31, half = lane >> 4, n16 = lane & 15;

  v8f acc[4];
  for (int g = 0; g < 4; ++g) for (int r = 0; r < 8; ++r) acc[g][r] = 0.f;
  for (int k0 = 0; k0 < HID; k0 += 32) {
    v16h a = load_a16h(hin, HID, bt * 16, NB - 1, k0);
#pragma unroll
    for (int g = 0; g < 4; ++g) {
      v16h b = load_bw16h(Whh, HID, g * HID + jt * 16, G4 - 1, k0);
      acc[g] = wmma16(a, b, acc[g]);
    }
  }
#pragma unroll
  for (int r = 0; r < 8; ++r) {
    int m = bt * 16 + r + 8 * half;      // batch index
    int j = jt * 16 + n16;               // hidden index
    size_t xb = ((size_t)tt * NB + m) * G4;
    float gi = acc[0][r] + xg[xb + j]           + bhh[j];
    float gf = acc[1][r] + xg[xb + HID + j]     + bhh[HID + j];
    float gg = acc[2][r] + xg[xb + 2 * HID + j] + bhh[2 * HID + j];
    float go = acc[3][r] + xg[xb + 3 * HID + j] + bhh[3 * HID + j];
    float cold = cst[m * HID + j];
    float cn = sigm(gf) * cold + sigm(gi) * tanhf(gg);
    float hn = sigm(go) * tanhf(cn);
    cst[m * HID + j]  = cn;
    hout[m * HID + j] = (_Float16)hn;
    xcat16[((size_t)tt * NB + m) * NX + dir * HID + j] = (_Float16)hn;
  }
}

// ---------------------------------------------------------------------------
// K5: GRU input projection [M5 x NX] x [NX x G3] + bih, TDM-staged A
// ---------------------------------------------------------------------------
__global__ void __launch_bounds__(256)
k_gru_xg(const _Float16* __restrict__ xcat16, const _Float16* __restrict__ W16,
         const float* __restrict__ bias, float* __restrict__ xg) {
  __shared__ _Float16 atile[16 * NX];   // 32 KB
  const int wib = threadIdx.x >> 5;
  const int w0 = blockIdx.x * 8;        // 0..30712
  const int mt = w0 / 96, nt = (w0 % 96) + wib;
  const int lane = threadIdx.x & 31, half = lane >> 4, n16 = lane & 15;

  stage_tdm(atile, xcat16, NX, mt * 16);

  v8f acc; for (int r = 0; r < 8; ++r) acc[r] = 0.f;
  for (int k0 = 0; k0 < NX; k0 += 32) {
    v16h a = lds_a16(atile, NX, k0);
    v16h b = load_bw16h(W16, NX, nt * 16, G3 - 1, k0);
    acc = wmma16(a, b, acc);
  }
#pragma unroll
  for (int r = 0; r < 8; ++r) {
    int m = mt * 16 + r + 8 * half;
    int col = nt * 16 + n16;
    xg[(size_t)m * G3 + col] = acc[r] + bias[col];
  }
}

// ---------------------------------------------------------------------------
// K6: fused GRU step. 3 WMMA accumulators = gates r,z,n.
// ---------------------------------------------------------------------------
__global__ void __launch_bounds__(256)
k_gru_step(const float* __restrict__ xg, const _Float16* __restrict__ Whh16,
           const float* __restrict__ bhh,
           const float* __restrict__ hin, float* __restrict__ hout,
           const _Float16* __restrict__ hin16, _Float16* __restrict__ hout16,
           _Float16* __restrict__ gout16, int s) {
  int w = blockIdx.x * 8 + (threadIdx.x >> 5);   // 0..255
  int bt = w >> 5, jt = w & 31;
  int lane = threadIdx.x & 31, half = lane >> 4, n16 = lane & 15;

  v8f acc[3];
  for (int g = 0; g < 3; ++g) for (int r = 0; r < 8; ++r) acc[g][r] = 0.f;
  for (int k0 = 0; k0 < NC; k0 += 32) {
    v16h a = load_a16h(hin16, NC, bt * 16, NB - 1, k0);
#pragma unroll
    for (int g = 0; g < 3; ++g) {
      v16h b = load_bw16h(Whh16, NC, g * NC + jt * 16, G3 - 1, k0);
      acc[g] = wmma16(a, b, acc[g]);
    }
  }
#pragma unroll
  for (int r = 0; r < 8; ++r) {
    int m = bt * 16 + r + 8 * half;
    int j = jt * 16 + n16;
    size_t xb = ((size_t)s * NB + m) * G3;
    float rg = sigm(xg[xb + j]          + acc[0][r] + bhh[j]);
    float zg = sigm(xg[xb + NC + j]     + acc[1][r] + bhh[NC + j]);
    float hn = acc[2][r] + bhh[2 * NC + j];
    float ng = tanhf(xg[xb + 2 * NC + j] + rg * hn);
    float ho = hin[m * NC + j];
    float hnew = (1.f - zg) * ng + zg * ho;
    hout[m * NC + j]   = hnew;
    hout16[m * NC + j] = (_Float16)hnew;
    gout16[((size_t)s * NB + m) * NC + j] = (_Float16)hnew;
  }
}

// ---------------------------------------------------------------------------
// K7: generator GEMM + ragged pack. N tiles padded to 320 so each block's 8
// waves share one M-tile (TDM-staged A). Valid rows -> packed offset off[b]+t.
// ---------------------------------------------------------------------------
__global__ void __launch_bounds__(256)
k_gen(const _Float16* __restrict__ gout16, const _Float16* __restrict__ genW16,
      const float* __restrict__ genb, const int* __restrict__ tl,
      const int* __restrict__ offs, float* __restrict__ out) {
  __shared__ _Float16 atile[16 * NC];   // 16 KB
  const int wib = threadIdx.x >> 5;
  const int w0 = blockIdx.x * 8;        // 0..102392
  const int mt = w0 / NTL, nt = (w0 % NTL) + wib;
  const int lane = threadIdx.x & 31, half = lane >> 4, n16 = lane & 15;

  stage_tdm(atile, gout16, NC, mt * 16);
  if (nt >= 313) return;               // dead padding waves (past the barrier)

  v8f acc; for (int r = 0; r < 8; ++r) acc[r] = 0.f;
  for (int k0 = 0; k0 < NC; k0 += 32) {
    v16h a = lds_a16(atile, NC, k0);
    v16h b = load_bw16h(genW16, NC, nt * 16, NCL - 1, k0);
    acc = wmma16(a, b, acc);
  }
#pragma unroll
  for (int r = 0; r < 8; ++r) {
    int m = mt * 16 + r + 8 * half;
    int t = m >> 7, bi = m & 127;
    int n = nt * 16 + n16;
    if (t < tl[bi] && n < NCL)
      out[((size_t)(offs[bi] + t)) * NCL + n] = acc[r] + genb[n];
  }
}

// ---------------------------------------------------------------------------
// Host launcher: all on `stream`, graph-capture safe.
// ---------------------------------------------------------------------------
extern "C" void kernel_launch(void* const* d_in, const int* in_sizes, int n_in,
                              void* d_out, int out_size, void* d_ws, size_t ws_size,
                              hipStream_t stream) {
  const float* feature = (const float*)d_in[0];
  const float* A       = (const float*)d_in[1];
  const int*   text    = (const int*)d_in[2];
  const int*   tl      = (const int*)d_in[3];
  const float* Wih_f = (const float*)d_in[4];
  const float* Whh_f = (const float*)d_in[5];
  const float* bih_f = (const float*)d_in[6];
  const float* bhh_f = (const float*)d_in[7];
  const float* Wih_b = (const float*)d_in[8];
  const float* Whh_b = (const float*)d_in[9];
  const float* bih_b = (const float*)d_in[10];
  const float* bhh_b = (const float*)d_in[11];
  const float* gWih  = (const float*)d_in[12];
  const float* gWhh  = (const float*)d_in[13];
  const float* gbih  = (const float*)d_in[14];
  const float* gbhh  = (const float*)d_in[15];
  const float* genW  = (const float*)d_in[16];
  const float* genb  = (const float*)d_in[17];
  const float* cemb  = (const float*)d_in[18];
  float* out = (float*)d_out;
  const int total = in_sizes[2];   // flat text length = #output rows

  char* ws = (char*)d_ws;
  size_t woff = 0;
  auto walloc = [&](size_t bytes) -> void* {
    void* p = ws + woff;
    woff += (bytes + 255) & ~(size_t)255;
    return p;
  };
  float* Sinv  = (float*)walloc((size_t)M5 * 4);
  int*   offs  = (int*)walloc((size_t)NB * 4);
  _Float16* Wih16f = (_Float16*)walloc((size_t)G4 * NC * 2);
  _Float16* Wih16b = (_Float16*)walloc((size_t)G4 * NC * 2);
  _Float16* Whh16f = (_Float16*)walloc((size_t)G4 * HID * 2);
  _Float16* Whh16b = (_Float16*)walloc((size_t)G4 * HID * 2);
  _Float16* gWih16 = (_Float16*)walloc((size_t)G3 * NX * 2);
  _Float16* gWhh16 = (_Float16*)walloc((size_t)G3 * NC * 2);
  _Float16* genW16 = (_Float16*)walloc((size_t)NCL * NC * 2);
  _Float16* Cpool16 = (_Float16*)walloc((size_t)M5 * NC * 2);
  _Float16* xcat16  = (_Float16*)walloc((size_t)M5 * NX * 2);
  _Float16* gout16  = (_Float16*)walloc((size_t)M5 * NC * 2);
  float* xgf = (float*)walloc((size_t)M5 * G4 * 4);
  float* xgb = (float*)walloc((size_t)M5 * G4 * 4);
  float* xgg = (float*)walloc((size_t)M5 * G3 * 4);
  const size_t stateBytes =
      4 * (size_t)NB * HID * 2 + 2 * (size_t)NB * HID * 4 +
      2 * (size_t)NB * NC * 4 + 2 * (size_t)NB * NC * 2;
  char* state = (char*)walloc(stateBytes);
  _Float16* hf16_0 = (_Float16*)state;
  _Float16* hf16_1 = hf16_0 + NB * HID;
  _Float16* hb16_0 = hf16_1 + NB * HID;
  _Float16* hb16_1 = hb16_0 + NB * HID;
  float* cf  = (float*)(hb16_1 + NB * HID);
  float* cb  = cf + NB * HID;
  float* hg0 = cb + NB * HID;
  float* hg1 = hg0 + NB * NC;
  _Float16* hg16_0 = (_Float16*)(hg1 + NB * NC);
  _Float16* hg16_1 = hg16_0 + NB * NC;

  hipMemsetAsync(state, 0, stateBytes, stream);

  k_cvt<<<dim3(256), 256, 0, stream>>>(Wih_f, Wih16f, G4 * NC / 4);
  k_cvt<<<dim3(256), 256, 0, stream>>>(Wih_b, Wih16b, G4 * NC / 4);
  k_cvt<<<dim3(256), 256, 0, stream>>>(Whh_f, Whh16f, G4 * HID / 4);
  k_cvt<<<dim3(256), 256, 0, stream>>>(Whh_b, Whh16b, G4 * HID / 4);
  k_cvt<<<dim3(256), 256, 0, stream>>>(gWih, gWih16, G3 * NX / 4);
  k_cvt<<<dim3(256), 256, 0, stream>>>(gWhh, gWhh16, G3 * NC / 4);
  k_cvt<<<dim3(256), 256, 0, stream>>>(genW, genW16, NCL * NC / 4);

  k_att_sums<<<dim3(M5 / 8), 256, 0, stream>>>(A, Sinv);
  k_offsets<<<dim3(1), 1, 0, stream>>>(tl, offs);
  k_pool<<<dim3(1536), 256, 0, stream>>>(A, feature, Sinv, Cpool16);
  k_lstm_xg<<<dim3(5120), 256, 0, stream>>>(Cpool16, Wih16f, bih_f,
                                            Wih16b, bih_b, xgf, xgb);
  k_emb<<<dim3(M5), 256, 0, stream>>>(text, tl, offs, cemb, xcat16, total);

  for (int s = 0; s < NT; ++s) {
    const _Float16* hfin = (s & 1) ? hf16_1 : hf16_0;
    _Float16* hfout      = (s & 1) ? hf16_0 : hf16_1;
    const _Float16* hbin = (s & 1) ? hb16_1 : hb16_0;
    _Float16* hbout      = (s & 1) ? hb16_0 : hb16_1;
    k_lstm_step<<<dim3(16, 2), 256, 0, stream>>>(xgf, xgb, Whh16f, Whh16b,
                                                 bhh_f, bhh_b,
                                                 hfin, hfout, cf,
                                                 hbin, hbout, cb, xcat16, s);
  }

  k_gru_xg<<<dim3(3840), 256, 0, stream>>>(xcat16, gWih16, gbih, xgg);

  for (int s = 0; s < NT; ++s) {
    const float* hin = (s & 1) ? hg1 : hg0;
    float* hout      = (s & 1) ? hg0 : hg1;
    const _Float16* hin16 = (s & 1) ? hg16_1 : hg16_0;
    _Float16* hout16      = (s & 1) ? hg16_0 : hg16_1;
    k_gru_step<<<dim3(32), 256, 0, stream>>>(xgg, gWhh16, gbhh,
                                             hin, hout, hin16, hout16,
                                             gout16, s);
  }

  k_gen<<<dim3(12800), 256, 0, stream>>>(gout16, genW16, genb, tl, offs, out);
  (void)n_in; (void)out_size; (void)ws_size;
}
